// GATEncoder_28853590295051
// MI455X (gfx1250) — matrix operand order repaired and checked by reading, not compiled, not run on hardware
//
#include <hip/hip_runtime.h>
#include <hip/hip_bf16.h>

typedef __attribute__((ext_vector_type(16))) __bf16 v16bf;
typedef __attribute__((ext_vector_type(8)))  __bf16 v8bf;
typedef __attribute__((ext_vector_type(4)))  __bf16 v4bf;
typedef __attribute__((ext_vector_type(8)))  float  v8f;

#define TSK 32   // K-step per WMMA (16x16x32 bf16)

// ---------------------------------------------------------------------------
// C[M x Nc] = A[M x K] (f32, row-major) * W[K x Nc] (f32, row-major) (+bias)
// via v_wmma_f32_16x16x32_bf16.
//   block = 256 thr = 8 wave32, block tile = 16 rows x 512 cols
//   each wave owns a 16x64 slab -> 4 WMMA per K-step sharing one A fragment
//   double-buffered LDS staging (A 2x1KB, B 2x32KB), one barrier / K-step
// GUARD=false: bounds-free fast path (M%16==0, Nc%512==0, K%32==0) with
// vectorized staging (float4 loads, ds_store_b64) -- covers the dominant
// layer-1 GEMM (10000x512x512). GUARD=true: scalar bounds-checked staging.
// ---------------------------------------------------------------------------
template <bool GUARD>
__global__ __launch_bounds__(256) void gemm_bf16_wmma(
    const float* __restrict__ A, const float* __restrict__ Wm,
    const float* __restrict__ bias, float* __restrict__ Cout,
    int M, int K, int Nc)
{
  __shared__ __align__(16) __bf16 As[2][16 * TSK];    // [buf][row][k]
  __shared__ __align__(16) __bf16 Bs[2][512 * TSK];   // [buf][col][k] (transposed)
  const int tid  = threadIdx.x;
  const int lane = tid & 31;
  const int wave = tid >> 5;
  const int row0 = blockIdx.y << 4;
  const int col0 = blockIdx.x << 9;
  const int wcol = wave << 6;                          // wave's 64-col slab

  auto stageA = [&](int k0, int buf) {
    if constexpr (!GUARD) {
      // one K-quad of one row per thread: float4 load -> 4x bf16 -> b64 store
      if (tid < 128) {
        const int r = tid >> 3, kq = (tid & 7) << 2;
        const float4 va = *(const float4*)&A[(size_t)(row0 + r) * K + k0 + kq];
        v4bf pk = {(__bf16)va.x, (__bf16)va.y, (__bf16)va.z, (__bf16)va.w};
        *(v4bf*)&As[buf][r * TSK + kq] = pk;
      }
    } else {
      for (int i = tid; i < 16 * TSK; i += 256) {
        int r = i >> 5, k = i & 31;
        int gr = row0 + r, gk = k0 + k;
        float v = (gr < M && gk < K) ? A[(size_t)gr * K + gk] : 0.f;
        As[buf][i] = (__bf16)v;
      }
    }
  };
  auto stageB = [&](int k0, int buf) {
    if constexpr (!GUARD) {
      // one K-quad of one column per thread: 4 coalesced b32 loads batched
      // before the convert chain, single ds_store_b64 (transposed stage)
#pragma unroll
      for (int i = tid; i < 512 * 8; i += 256) {       // 16 iters, const trip
        const int c  = i & 511;
        const int kq = (i >> 9) << 2;
        const float* p = &Wm[(size_t)(k0 + kq) * Nc + (col0 + c)];
        const float v0 = p[0];
        const float v1 = p[(size_t)Nc];
        const float v2 = p[(size_t)2 * Nc];
        const float v3 = p[(size_t)3 * Nc];
        v4bf pk = {(__bf16)v0, (__bf16)v1, (__bf16)v2, (__bf16)v3};
        *(v4bf*)&Bs[buf][c * TSK + kq] = pk;
      }
    } else {
      for (int i = tid; i < 512 * TSK; i += 256) {
        int k = i >> 9, c = i & 511;
        int gc = col0 + c, gk = k0 + k;
        float v = (gc < Nc && gk < K) ? Wm[(size_t)gk * Nc + gc] : 0.f;
        Bs[buf][c * TSK + k] = (__bf16)v;
      }
    }
  };

  v8f acc[4] = {};
  const int nk = (K + TSK - 1) / TSK;

  stageA(0, 0);
  stageB(0, 0);
  __syncthreads();

  for (int t = 0; t < nk; ++t) {
    const int buf = t & 1;
    if (t + 1 < nk) {                 // stage next tile into the other buffer
      stageA((t + 1) * TSK, buf ^ 1);
      stageB((t + 1) * TSK, buf ^ 1);
    }
    if (t + 2 < nk) {                 // pull tile t+2 toward L2 (global_prefetch_b8)
      __builtin_prefetch(&A[(size_t)(row0 + (tid & 15)) * K + (size_t)(t + 2) * TSK], 0, 1);
      __builtin_prefetch(&Wm[(size_t)((t + 2) * TSK + (tid >> 4)) * Nc + col0], 0, 1);
    }

    // A fragment per ISA 16-bit A 16x32 layout:
    //   elem j -> k = (j>>3)*16 + (lane>>4)*8 + (j&7), row = lane&15
    const int arow = lane & 15;
    const int asel = (lane >> 4) << 3;
    v8bf alo = *(const v8bf*)&As[buf][arow * TSK + asel];
    v8bf ahi = *(const v8bf*)&As[buf][arow * TSK + 16 + asel];
    v16bf afrag = __builtin_shufflevector(alo, ahi,
        0,1,2,3,4,5,6,7,8,9,10,11,12,13,14,15);

    // B fragments: elem j -> k = (j>>3)*16 + (lane>>4)*8 + (j&7), col = lane&15
    const int bk = (lane >> 4) << 3;
#pragma unroll
    for (int tt = 0; tt < 4; ++tt) {
      const int bcol = wcol + (tt << 4) + (lane & 15);
      v8bf blo = *(const v8bf*)&Bs[buf][bcol * TSK + bk];
      v8bf bhi = *(const v8bf*)&Bs[buf][bcol * TSK + 16 + bk];
      v16bf bfrag = __builtin_shufflevector(blo, bhi,
          0,1,2,3,4,5,6,7,8,9,10,11,12,13,14,15);
      acc[tt] = __builtin_amdgcn_wmma_f32_16x16x32_bf16(
          false, afrag, false, bfrag, (short)0, acc[tt], false, false);
    }
    __syncthreads();
  }

  // D layout: lane<16 -> M=r, N=lane ; lane>=16 -> M=8+r, N=lane-16
  const int mb = row0 + ((lane >> 4) << 3);
#pragma unroll
  for (int tt = 0; tt < 4; ++tt) {
    const int n = col0 + wcol + (tt << 4) + (lane & 15);
    if (!GUARD || n < Nc) {
      const float bv = bias ? bias[n] : 0.f;
#pragma unroll
      for (int r = 0; r < 8; ++r) {
        const int m = mb + r;
        if (!GUARD || m < M) Cout[(size_t)m * Nc + n] = acc[tt][r] + bv;
      }
    }
  }
}

// ---------------------------------------------------------------------------
__global__ void fillk(float* __restrict__ p, float v, int n) {
  int i = blockIdx.x * blockDim.x + threadIdx.x;
  if (i < n) p[i] = v;
}

// src/dst with self-loops appended
__global__ void build_edges(const int* __restrict__ ei, int E, int N,
                            int* __restrict__ src, int* __restrict__ dst) {
  int i = blockIdx.x * blockDim.x + threadIdx.x;
  int Et = E + N;
  if (i >= Et) return;
  if (i < E) { src[i] = ei[i]; dst[i] = ei[E + i]; }
  else       { src[i] = i - E; dst[i] = i - E; }
}

// per-node attention logits: as[n,h] = sum_c h[n,h,c]*a_src[h,c] (same for ad)
__global__ void node_attn(const float* __restrict__ h,
                          const float* __restrict__ aw_s,
                          const float* __restrict__ aw_d,
                          float* __restrict__ asn, float* __restrict__ adn,
                          int N, int H, int C) {
  int i = blockIdx.x * blockDim.x + threadIdx.x;
  if (i >= N * H) return;
  int n = i / H, hh = i % H;
  const float* hp = h + (size_t)n * H * C + (size_t)hh * C;
  float s = 0.f, d = 0.f;
  for (int c = 0; c < C; ++c) {
    float v = hp[c];
    s += v * aw_s[hh * C + c];
    d += v * aw_d[hh * C + c];
  }
  asn[i] = s; adn[i] = d;
}

__device__ __forceinline__ float leaky02(float v) {
  return v > 0.f ? v : 0.2f * v;
}

__device__ __forceinline__ void atomicMaxF32(float* addr, float v) {
  unsigned* u = (unsigned*)addr;
  unsigned old = *u;
  float ov = __uint_as_float(old);
  while (v > ov) {
    unsigned assumed = old;
    old = atomicCAS(u, assumed, __float_as_uint(v));
    if (old == assumed) break;
    ov = __uint_as_float(old);
  }
}

// pass 1: segment max over incoming edges of dst
__global__ void edge_max(const int* __restrict__ src, const int* __restrict__ dst,
                         const float* __restrict__ asn, const float* __restrict__ adn,
                         float* __restrict__ mx, int Et, int H) {
  int i = blockIdx.x * blockDim.x + threadIdx.x;
  if (i >= Et * H) return;
  int e = i / H, hh = i % H;
  float v = leaky02(asn[src[e] * H + hh] + adn[dst[e] * H + hh]);
  atomicMaxF32(&mx[dst[e] * H + hh], v);
}

// pass 2: ex = exp(e - m[dst]); den[dst] += ex
__global__ void edge_expsum(const int* __restrict__ src, const int* __restrict__ dst,
                            const float* __restrict__ asn, const float* __restrict__ adn,
                            const float* __restrict__ mx, float* __restrict__ ex,
                            float* __restrict__ den, int Et, int H) {
  int i = blockIdx.x * blockDim.x + threadIdx.x;
  if (i >= Et * H) return;
  int e = i / H, hh = i % H;
  int d = dst[e];
  float v = leaky02(asn[src[e] * H + hh] + adn[d * H + hh]);
  float xv = __expf(v - mx[d * H + hh]);
  ex[i] = xv;
  atomicAdd(&den[d * H + hh], xv);
}

// pass 3: agg[dst] += alpha * h[src]; one thread per (edge, head, 4 channels)
__global__ void edge_aggr(const int* __restrict__ src, const int* __restrict__ dst,
                          const float* __restrict__ ex, const float* __restrict__ den,
                          const float* __restrict__ h, float* __restrict__ agg,
                          int Et, int H, int C) {
  int i = blockIdx.x * blockDim.x + threadIdx.x;
  int c4n = C >> 2;
  int per = H * c4n;
  if (i >= Et * per) return;
  int e   = i / per;
  int rem = i % per;
  int hh  = rem / c4n;
  int c0  = (rem % c4n) << 2;
  int s = src[e], d = dst[e];
  float alpha = ex[(size_t)e * H + hh] / (den[d * H + hh] + 1e-16f);
  const float4 hv = *(const float4*)&h[(size_t)s * H * C + (size_t)hh * C + c0];
  float* op = &agg[(size_t)d * H * C + (size_t)hh * C + c0];
  atomicAdd(op + 0, alpha * hv.x);
  atomicAdd(op + 1, alpha * hv.y);
  atomicAdd(op + 2, alpha * hv.z);
  atomicAdd(op + 3, alpha * hv.w);
}

// (agg + b) -> BN(eval) -> ELU
__global__ void bn_elu(const float* __restrict__ agg, const float* __restrict__ b,
                       const float* __restrict__ g, const float* __restrict__ be,
                       const float* __restrict__ rm, const float* __restrict__ rv,
                       float* __restrict__ out, int N, int F) {
  int i = blockIdx.x * blockDim.x + threadIdx.x;
  if (i >= N * F) return;
  int f = i % F;
  float v = agg[i] + b[f];
  v = (v - rm[f]) * (g[f] * __frsqrt_rn(rv[f] + 1e-5f)) + be[f];
  out[i] = v > 0.f ? v : (__expf(v) - 1.f);
}

__global__ void pool_accum(const float* __restrict__ h, const int* __restrict__ batch,
                           float* __restrict__ sums, float* __restrict__ cnt,
                           int N, int F) {
  int i = blockIdx.x * blockDim.x + threadIdx.x;
  if (i >= N * F) return;
  int n = i / F, f = i % F;
  int bb = batch[n];
  atomicAdd(&sums[(size_t)bb * F + f], h[i]);
  if (f == 0) atomicAdd(&cnt[bb], 1.f);
}

__global__ void pool_div(float* __restrict__ sums, const float* __restrict__ cnt,
                         int Bn, int F) {
  int i = blockIdx.x * blockDim.x + threadIdx.x;
  if (i >= Bn * F) return;
  sums[i] /= fmaxf(cnt[i / F], 1.f);
}

// ---------------------------------------------------------------------------
static inline void launch_gemm(const float* A, const float* Wm, const float* bias,
                               float* Cout, int M, int K, int Nc, hipStream_t stream) {
  dim3 gg((Nc + 511) / 512, (M + 15) / 16);
  const bool fast = (M % 16 == 0) && (Nc % 512 == 0) && (K % TSK == 0);
  if (fast)
    gemm_bf16_wmma<false><<<gg, 256, 0, stream>>>(A, Wm, bias, Cout, M, K, Nc);
  else
    gemm_bf16_wmma<true><<<gg, 256, 0, stream>>>(A, Wm, bias, Cout, M, K, Nc);
}

extern "C" void kernel_launch(void* const* d_in, const int* in_sizes, int n_in,
                              void* d_out, int out_size, void* d_ws, size_t ws_size,
                              hipStream_t stream) {
  const int IN = 5, HID = 128, HEADS = 4;
  const float* x   = (const float*)d_in[0];
  const int* ei    = (const int*)d_in[1];
  const int* batch = (const int*)d_in[2];
  const int N    = in_sizes[0] / IN;
  const int E    = in_sizes[1] / 2;
  const int Etot = E + N;
  const int EMB  = in_sizes[28];
  const int Bn   = out_size / EMB;

  const float* Wl[3] = {(const float*)d_in[3],  (const float*)d_in[11], (const float*)d_in[19]};
  const float* Aws[3]= {(const float*)d_in[4],  (const float*)d_in[12], (const float*)d_in[20]};
  const float* Awd[3]= {(const float*)d_in[5],  (const float*)d_in[13], (const float*)d_in[21]};
  const float* bl[3] = {(const float*)d_in[6],  (const float*)d_in[14], (const float*)d_in[22]};
  const float* gl[3] = {(const float*)d_in[7],  (const float*)d_in[15], (const float*)d_in[23]};
  const float* bel[3]= {(const float*)d_in[8],  (const float*)d_in[16], (const float*)d_in[24]};
  const float* rml[3]= {(const float*)d_in[9],  (const float*)d_in[17], (const float*)d_in[25]};
  const float* rvl[3]= {(const float*)d_in[10], (const float*)d_in[18], (const float*)d_in[26]};
  const float* embW  = (const float*)d_in[27];
  const float* embb  = (const float*)d_in[28];
  (void)n_in; (void)ws_size;

  // ---- carve workspace (3 rotating N*512 feature buffers + edge scratch) ----
  char* w = (char*)d_ws;
  auto carve = [&](size_t bytes) -> char* {
    char* p = w; w += (bytes + 255) & ~(size_t)255; return p;
  };
  int*   src2 = (int*)carve((size_t)Etot * 4);
  int*   dst2 = (int*)carve((size_t)Etot * 4);
  float* P0   = (float*)carve((size_t)N * 512 * 4);  // features (bn output)
  float* P1   = (float*)carve((size_t)N * 512 * 4);  // linear output h
  float* P2   = (float*)carve((size_t)N * 512 * 4);  // aggregation accumulator
  float* asn  = (float*)carve((size_t)N * HEADS * 4);
  float* adn  = (float*)carve((size_t)N * HEADS * 4);
  float* mx   = (float*)carve((size_t)N * HEADS * 4);
  float* den  = (float*)carve((size_t)N * HEADS * 4);
  float* ex   = (float*)carve((size_t)Etot * HEADS * 4);
  float* sums = (float*)carve((size_t)Bn * HID * 4);
  float* cnt  = (float*)carve((size_t)Bn * 4);

  build_edges<<<(Etot + 255) / 256, 256, 0, stream>>>(ei, E, N, src2, dst2);

  const int Hh[3]  = {HEADS, HEADS, 1};
  const int Din[3] = {IN, HEADS * HID, HEADS * HID};
  const float* fin = x;

  for (int L = 0; L < 3; ++L) {
    const int H = Hh[L], C = HID, K = Din[L], F = H * C;
    launch_gemm(fin, Wl[L], nullptr, P1, N, K, F, stream);

    const int nh = N * H;
    node_attn<<<(nh + 255) / 256, 256, 0, stream>>>(P1, Aws[L], Awd[L], asn, adn, N, H, C);
    fillk<<<(nh + 255) / 256, 256, 0, stream>>>(mx, -1e30f, nh);
    fillk<<<(nh + 255) / 256, 256, 0, stream>>>(den, 0.f, nh);
    const int nf = N * F;
    fillk<<<(nf + 255) / 256, 256, 0, stream>>>(P2, 0.f, nf);

    const int eh = Etot * H;
    edge_max<<<(eh + 255) / 256, 256, 0, stream>>>(src2, dst2, asn, adn, mx, Etot, H);
    edge_expsum<<<(eh + 255) / 256, 256, 0, stream>>>(src2, dst2, asn, adn, mx, ex, den, Etot, H);
    const int tot = Etot * H * (C / 4);
    edge_aggr<<<(tot + 255) / 256, 256, 0, stream>>>(src2, dst2, ex, den, P1, P2, Etot, H, C);

    bn_elu<<<(nf + 255) / 256, 256, 0, stream>>>(P2, bl[L], gl[L], bel[L], rml[L], rvl[L],
                                                 P0, N, F);
    fin = P0;
  }

  // global mean pool + final projection (WMMA again: 64x128 @ 128x64 + bias)
  fillk<<<(Bn * HID + 255) / 256, 256, 0, stream>>>(sums, 0.f, Bn * HID);
  fillk<<<(Bn + 255) / 256, 256, 0, stream>>>(cnt, 0.f, Bn);
  pool_accum<<<(N * HID + 255) / 256, 256, 0, stream>>>(P0, batch, sums, cnt, N, HID);
  pool_div<<<(Bn * HID + 255) / 256, 256, 0, stream>>>(sums, cnt, Bn, HID);

  launch_gemm(sums, embW, embb, (float*)d_out, Bn, HID, EMB, stream);
}
